// MultilayerGRU_29901562315292
// MI455X (gfx1250) — compile-verified, matrix operand order checked by reading, not run on hardware
//
#include <hip/hip_runtime.h>
#include <hip/hip_bf16.h>
#include <cstdint>
#include <cstddef>

#define B__  32
#define S__  2048
#define I__  128
#define H__  512
#define O__  128
#define BT   16      // batch rows per workgroup (one WMMA M tile)
#define NTH  512     // 16 waves of 32
#define HP   (H__ + 8)   // padded bf16 row stride (even -> 4B aligned pairs)
#define IP   (I__ + 8)

typedef __attribute__((ext_vector_type(16))) __bf16 v16bf;
typedef __attribute__((ext_vector_type(8)))  float  v8f;

__device__ __forceinline__ float sigmoidf_(float x) { return 1.0f / (1.0f + __expf(-x)); }

// ---------------------------------------------------------------------------
// Weight repack: f32 (K,N) row-major  ->  WMMA-B bf16 tiles.
// Tile (kt,nt): 32 lanes x 16 halves. lane<16 -> N = nt*16+lane, K pairs
// {0..7,16..23}; lane>=16 -> same N set, K pairs {8..15,24..31} (ISA 7.12.2).
// dst element index == ((kt*NT+nt)*32 + lane)*16 + j
// ---------------------------------------------------------------------------
__global__ void pack_b_tiles(const float* __restrict__ src, __bf16* __restrict__ dst,
                             int K, int N) {
  const int NT = N >> 4;
  const long total = (long)K * (long)N;
  long tid = (long)blockIdx.x * blockDim.x + threadIdx.x;
  if (tid >= total) return;
  long idx = tid;
  int j    = (int)(idx & 15); idx >>= 4;
  int lane = (int)(idx & 31); idx >>= 5;
  int nt   = (int)(idx % NT);
  int kt   = (int)(idx / NT);
  int n  = nt * 16 + (lane & 15);
  int p  = j >> 1, h = j & 1;
  int kl = (p < 4 ? 2 * p : 16 + 2 * (p - 4)) + h + ((lane >> 4) << 3);
  int k  = kt * 32 + kl;
  dst[tid] = (__bf16)src[(long)k * N + n];
}

// ---------------------------------------------------------------------------
struct GruP {
  const float*  x;
  const float*  h0;
  const __bf16* pWx[2][3];    // z,r,g per layer (packed B tiles)
  const __bf16* pWh[2][3];
  const float*  bias[2][3];
  const __bf16* pWy;
  const float*  by;
  float*        out;
};

__device__ __forceinline__ void load_A(uint32_t* Au, const __bf16* aRow,
                                       int kbase, int khi) {
#pragma unroll
  for (int p = 0; p < 8; ++p) {
    int kl = (p < 4 ? 2 * p : 16 + 2 * (p - 4)) + khi;
    Au[p] = *(const uint32_t*)(aRow + kbase + kl);   // bf16 pair from LDS
  }
}

// Two N-tiles sharing one A operand, with software-pipelined (double-buffered)
// B loads: next k-step's 4x global_load_b128 are in flight while the current
// two WMMAs execute, so waits become partial instead of loadcnt==0.
__device__ __forceinline__ void gemm_pair(v8f& c0, v8f& c1,
    const __bf16* aBase, int aStride, int ksteps,
    const __bf16* bt0, const __bf16* bt1, int ktStrideU4, int lane) {
  const __bf16* aRow = aBase + (size_t)(lane & 15) * aStride;
  const int khi = (lane >> 4) << 3;
  const uint4* b0p = (const uint4*)bt0 + lane * 2;
  const uint4* b1p = (const uint4*)bt1 + lane * 2;

  union { v16bf v; uint4 q[2]; } B0c, B1c, B0n, B1n;
  B0c.q[0] = b0p[0]; B0c.q[1] = b0p[1];
  B1c.q[0] = b1p[0]; B1c.q[1] = b1p[1];

  for (int ks = 0; ks < ksteps; ++ks) {
    if (ks + 1 < ksteps) {                       // prefetch next B tiles
      const uint4* n0 = b0p + (size_t)(ks + 1) * ktStrideU4;
      const uint4* n1 = b1p + (size_t)(ks + 1) * ktStrideU4;
      B0n.q[0] = n0[0]; B0n.q[1] = n0[1];
      B1n.q[0] = n1[0]; B1n.q[1] = n1[1];
    }
    union { v16bf v; uint32_t u[8]; } A;
    load_A(A.u, aRow, ks * 32, khi);
    c0 = __builtin_amdgcn_wmma_f32_16x16x32_bf16(false, A.v, false, B0c.v,
                                                 (short)0, c0, false, false);
    c1 = __builtin_amdgcn_wmma_f32_16x16x32_bf16(false, A.v, false, B1c.v,
                                                 (short)0, c1, false, false);
    B0c = B0n; B1c = B1n;
  }
}

// single-tile variant (output head), same B double-buffering
__device__ __forceinline__ void gemm_one(v8f& c0,
    const __bf16* aBase, int aStride, int ksteps,
    const __bf16* bt0, int ktStrideU4, int lane) {
  const __bf16* aRow = aBase + (size_t)(lane & 15) * aStride;
  const int khi = (lane >> 4) << 3;
  const uint4* b0p = (const uint4*)bt0 + lane * 2;
  union { v16bf v; uint4 q[2]; } B0c, B0n;
  B0c.q[0] = b0p[0]; B0c.q[1] = b0p[1];
  for (int ks = 0; ks < ksteps; ++ks) {
    if (ks + 1 < ksteps) {
      const uint4* n0 = b0p + (size_t)(ks + 1) * ktStrideU4;
      B0n.q[0] = n0[0]; B0n.q[1] = n0[1];
    }
    union { v16bf v; uint32_t u[8]; } A;
    load_A(A.u, aRow, ks * 32, khi);
    c0 = __builtin_amdgcn_wmma_f32_16x16x32_bf16(false, A.v, false, B0c.v,
                                                 (short)0, c0, false, false);
    B0c = B0n;
  }
}

__global__ __launch_bounds__(NTH) void gru_persistent(GruP P) {
  __shared__ __align__(16) float  hS[2][BT][H__];   // f32 master h per layer
  __shared__ __align__(16) __bf16 hB[2][BT][HP];    // bf16 GEMM copies
  __shared__ __align__(16) __bf16 xB[BT][IP];       // x_t slab (bf16)
  __shared__ __align__(16) __bf16 rhB[BT][HP];      // r*h (bf16)
  __shared__ __align__(16) float  zS[BT][H__];
  __shared__ __align__(16) float  rS[BT][H__];

  const int tid   = threadIdx.x;
  const int lane  = tid & 31;
  const int wave  = tid >> 5;
  const int bbase = blockIdx.x * BT;
  const int kt4_32 = 32 * 64;   // uint4 per k-step, NT=32 (32 tiles * 1KB / 16B)
  const int kt4_8  = 8 * 64;    // uint4 per k-step, NT=8 (Wy)

  // init h from h0 input, layout (B, L, H)
  for (int idx = tid; idx < 2 * BT * H__; idx += NTH) {
    int col = idx % H__;
    int row = (idx / H__) % BT;
    int l   = idx / (H__ * BT);
    float v = P.h0[(((size_t)(bbase + row)) * 2 + l) * H__ + col];
    hS[l][row][col] = v;
    hB[l][row][col] = (__bf16)v;
  }
  __syncthreads();

  for (int t = 0; t < S__; ++t) {
    // stage x_t (BT x I) into LDS as bf16
    for (int idx = tid; idx < BT * I__; idx += NTH) {
      int col = idx % I__;
      int row = idx / I__;
      xB[row][col] = (__bf16)P.x[(((size_t)(bbase + row)) * S__ + t) * I__ + col];
    }
    if (t + 1 < S__ && tid < BT)   // pull next timestep into cache early
      __builtin_prefetch(&P.x[(((size_t)(bbase + tid)) * S__ + (t + 1)) * I__], 0, 0);
    __syncthreads();

#pragma unroll
    for (int l = 0; l < 2; ++l) {
      const __bf16* inp  = (l == 0) ? &xB[0][0] : &hB[0][0][0];
      const int inStride = (l == 0) ? IP : HP;
      const int kstepsA  = (l == 0) ? (I__ / 32) : (H__ / 32);

      // ---- phase 1: z & r as pairs (shared A); 2 column groups / wave ----
#pragma unroll
      for (int q = 0; q < 2; ++q) {
        int nt  = wave + 16 * q;
        int col = nt * 16 + (lane & 15);
        float bz = P.bias[l][0][col];
        float br = P.bias[l][1][col];
        v8f cz = {bz, bz, bz, bz, bz, bz, bz, bz};
        v8f cr = {br, br, br, br, br, br, br, br};
        gemm_pair(cz, cr, inp, inStride, kstepsA,
                  P.pWx[l][0] + (size_t)nt * 512, P.pWx[l][1] + (size_t)nt * 512,
                  kt4_32, lane);
        gemm_pair(cz, cr, &hB[l][0][0], HP, H__ / 32,
                  P.pWh[l][0] + (size_t)nt * 512, P.pWh[l][1] + (size_t)nt * 512,
                  kt4_32, lane);
        int r0 = (lane >> 4) * 8;
#pragma unroll
        for (int v = 0; v < 8; ++v) {
          zS[r0 + v][col] = sigmoidf_(cz[v]);
          rS[r0 + v][col] = sigmoidf_(cr[v]);
        }
      }
      __syncthreads();

      // ---- r * h_prev -> bf16 ----
      for (int idx = tid; idx < BT * H__; idx += NTH) {
        int col = idx % H__;
        int row = idx / H__;
        rhB[row][col] = (__bf16)(rS[row][col] * hS[l][row][col]);
      }
      __syncthreads();

      // ---- phase 2: two g tiles as a pair (shared A); fused h update ----
      {
        int nt0 = wave, nt1 = wave + 16;
        int col0 = nt0 * 16 + (lane & 15);
        int col1 = nt1 * 16 + (lane & 15);
        float b0 = P.bias[l][2][col0];
        float b1 = P.bias[l][2][col1];
        v8f c0 = {b0, b0, b0, b0, b0, b0, b0, b0};
        v8f c1 = {b1, b1, b1, b1, b1, b1, b1, b1};
        gemm_pair(c0, c1, inp, inStride, kstepsA,
                  P.pWx[l][2] + (size_t)nt0 * 512, P.pWx[l][2] + (size_t)nt1 * 512,
                  kt4_32, lane);
        gemm_pair(c0, c1, &rhB[0][0], HP, H__ / 32,
                  P.pWh[l][2] + (size_t)nt0 * 512, P.pWh[l][2] + (size_t)nt1 * 512,
                  kt4_32, lane);
        int r0 = (lane >> 4) * 8;
#pragma unroll
        for (int v = 0; v < 8; ++v) {
          float g  = tanhf(c0[v]);
          float z  = zS[r0 + v][col0];
          float h  = hS[l][r0 + v][col0];
          float hn = z * h + (1.0f - z) * g;
          hS[l][r0 + v][col0] = hn;
          hB[l][r0 + v][col0] = (__bf16)hn;

          g  = tanhf(c1[v]);
          z  = zS[r0 + v][col1];
          h  = hS[l][r0 + v][col1];
          hn = z * h + (1.0f - z) * g;
          hS[l][r0 + v][col1] = hn;
          hB[l][r0 + v][col1] = (__bf16)hn;
        }
      }
      __syncthreads();
    }

    // ---- output head: y_t = h1_t @ Wy + by ; 8 N-tiles on waves 0..7 ----
    if (wave < 8) {
      int nt  = wave;
      int col = nt * 16 + (lane & 15);
      float b = P.by[col];
      v8f c = {b, b, b, b, b, b, b, b};
      gemm_one(c, &hB[1][0][0], HP, H__ / 32,
               P.pWy + (size_t)nt * 512, kt4_8, lane);
      int r0 = (lane >> 4) * 8;
#pragma unroll
      for (int v = 0; v < 8; ++v)
        P.out[(((size_t)(bbase + r0 + v)) * S__ + t) * O__ + col] = c[v];
    }
    __syncthreads();
  }

  // final hidden_state (B, L, H) appended after layer_output
  float* hsOut = P.out + (size_t)B__ * S__ * O__;
  for (int idx = tid; idx < 2 * BT * H__; idx += NTH) {
    int col = idx % H__;
    int row = (idx / H__) % BT;
    int l   = idx / (H__ * BT);
    hsOut[(((size_t)(bbase + row)) * 2 + l) * H__ + col] = hS[l][row][col];
  }
}

// ---------------------------------------------------------------------------
extern "C" void kernel_launch(void* const* d_in, const int* in_sizes, int n_in,
                              void* d_out, int out_size, void* d_ws, size_t ws_size,
                              hipStream_t stream) {
  (void)in_sizes; (void)n_in; (void)out_size; (void)ws_size;
  const float* x  = (const float*)d_in[0];
  const float* h0 = (const float*)d_in[1];

  // input indices: per layer {Wxz,Whz,bz, Wxr,Whr,br, Wxg,Whg,bg}
  const int wx_i[2][3] = {{2, 5, 8},  {11, 14, 17}};   // z,r,g
  const int wh_i[2][3] = {{3, 6, 9},  {12, 15, 18}};
  const int bi_i[2][3] = {{4, 7, 10}, {13, 16, 19}};

  __bf16* ws = (__bf16*)d_ws;
  size_t off = 0;
  auto alloc = [&](size_t halves) { __bf16* p = ws + off; off += halves; return p; };
  auto pack  = [&](const float* src, __bf16* dst, int K, int N) {
    long total = (long)K * N;
    int blocks = (int)((total + 255) / 256);
    hipLaunchKernelGGL(pack_b_tiles, dim3(blocks), dim3(256), 0, stream, src, dst, K, N);
  };

  GruP P{};
  P.x = x; P.h0 = h0; P.out = (float*)d_out;
  for (int l = 0; l < 2; ++l) {
    int Kx = (l == 0) ? I__ : H__;
    for (int g = 0; g < 3; ++g) {
      __bf16* px = alloc((size_t)Kx * H__);
      pack((const float*)d_in[wx_i[l][g]], px, Kx, H__);
      P.pWx[l][g] = px;
      __bf16* ph = alloc((size_t)H__ * H__);
      pack((const float*)d_in[wh_i[l][g]], ph, H__, H__);
      P.pWh[l][g] = ph;
      P.bias[l][g] = (const float*)d_in[bi_i[l][g]];
    }
  }
  __bf16* py = alloc((size_t)H__ * O__);
  pack((const float*)d_in[20], py, H__, O__);
  P.pWy = py;
  P.by  = (const float*)d_in[21];

  hipLaunchKernelGGL(gru_persistent, dim3(B__ / BT), dim3(NTH), 0, stream, P);
}
